// Attn_4209067950204
// MI455X (gfx1250) — matrix operand order repaired and checked by reading, not compiled, not run on hardware
//
#include <hip/hip_runtime.h>
#include <hip/hip_bf16.h>

// ---------------------------------------------------------------------------
// Problem constants (match reference)
// ---------------------------------------------------------------------------
#define STATE_LEN 2048
#define SEQ_LEN   4096
#define HIDDEN    2048

// ---------------------------------------------------------------------------
// CDNA5 WMMA vector types
// ---------------------------------------------------------------------------
typedef __attribute__((ext_vector_type(16))) __bf16 v16bf;
typedef __attribute__((ext_vector_type(8)))  __bf16 bf16x8;
typedef __attribute__((ext_vector_type(4)))  __bf16 bf16x4;
typedef __attribute__((ext_vector_type(8)))  float  v8f;

// ---------------------------------------------------------------------------
// f32 -> bf16 conversion (4 elems / thread, fully vectorized)
// ---------------------------------------------------------------------------
__global__ void cvt_f32_bf16(const float* __restrict__ in,
                             __bf16* __restrict__ out, int n4) {
    int i = blockIdx.x * blockDim.x + threadIdx.x;
    if (i < n4) {
        const float4 f = *(const float4*)(in + 4 * (size_t)i);
        bf16x4 o;
        o[0] = (__bf16)f.x; o[1] = (__bf16)f.y;
        o[2] = (__bf16)f.z; o[3] = (__bf16)f.w;
        *(bf16x4*)(out + 4 * (size_t)i) = o;
    }
}

// ---------------------------------------------------------------------------
// NT GEMM on WMMA:  C[M,N] = A[M,K] @ B[N,K]^T  (+ bias over N, optional)
// A, B: bf16 row-major, K contiguous.
// BIAS==true  -> C stored as bf16 with bias added (energies pass)
// BIAS==false -> C stored as f32 (scores pass)
//
// Wave tile: 64(M) x 64(N) = 4x4 WMMA 16x16x32 tiles
//            -> 16 WMMAs per 16 b128 loads per K-step (issue ratio 1.0)
// Block: 8 waves as 2(M) x 4(N) -> 128 x 256 per block.
// ---------------------------------------------------------------------------
template <bool BIAS>
__global__ __launch_bounds__(256)
void gemm_nt_wmma(const __bf16* __restrict__ A,
                  const __bf16* __restrict__ B,
                  const float* __restrict__ bias,
                  __bf16* __restrict__ Cbf,
                  float* __restrict__ Cf,
                  int M, int N, int K) {
    const int lane = threadIdx.x & 31;
    const int wave = threadIdx.x >> 5;
    const int wm   = wave & 1;          // 0..1 : M direction
    const int wn   = wave >> 1;         // 0..3 : N direction
    const int m0   = blockIdx.y * 128 + wm * 64;
    const int n0   = blockIdx.x * 256 + wn * 64;

    const int mr = lane & 15;           // row-within-tile carried by this lane
    const int kh = lane >> 4;           // K-half select (ISA fragment layout)

    // Per-lane base pointers (hoisted out of the K loop):
    //   A fragment 16x32: lane(m,kh) holds K = 8*kh+0..7  and  16+8*kh+0..7
    //   B fragment 32x16: lane n(=mr),kh holds column n,   K = 8*kh+0..7  and  16+8*kh+0..7
    const __bf16* pa[4];
    const __bf16* pb[4];
#pragma unroll
    for (int mi = 0; mi < 4; ++mi)
        pa[mi] = A + (size_t)(m0 + mi * 16 + mr) * K + 8 * kh;
#pragma unroll
    for (int ni = 0; ni < 4; ++ni)
        pb[ni] = B + (size_t)(n0 + ni * 16 + mr) * K + 8 * kh;

    v8f acc[4][4];
#pragma unroll
    for (int mi = 0; mi < 4; ++mi)
#pragma unroll
        for (int ni = 0; ni < 4; ++ni) acc[mi][ni] = (v8f){};

    for (int kb = 0; kb < K; kb += 32) {
        v16bf afrag[4];
#pragma unroll
        for (int mi = 0; mi < 4; ++mi) {
            const bf16x8 lo = *(const bf16x8*)(pa[mi] + kb);
            const bf16x8 hi = *(const bf16x8*)(pa[mi] + kb + 16);
#pragma unroll
            for (int e = 0; e < 8; ++e) { afrag[mi][e] = lo[e]; afrag[mi][e + 8] = hi[e]; }
        }
        v16bf bfrag[4];
#pragma unroll
        for (int ni = 0; ni < 4; ++ni) {
            const bf16x8 lo = *(const bf16x8*)(pb[ni] + kb);
            const bf16x8 hi = *(const bf16x8*)(pb[ni] + kb + 16);
#pragma unroll
            for (int e = 0; e < 8; ++e) { bfrag[ni][e] = lo[e]; bfrag[ni][e + 8] = hi[e]; }
        }
        // 16 WMMAs: A fragments reused across 4 N tiles, B across 4 M tiles
#pragma unroll
        for (int mi = 0; mi < 4; ++mi)
#pragma unroll
            for (int ni = 0; ni < 4; ++ni)
                acc[mi][ni] = __builtin_amdgcn_wmma_f32_16x16x32_bf16(
                    false, afrag[mi], false, bfrag[ni],
                    (short)0, acc[mi][ni], false, false);
    }

    // store: C/D layout: VGPR r -> M = r + 8*kh (lanes 0-15: +0, 16-31: +8), N = lane&15
#pragma unroll
    for (int mi = 0; mi < 4; ++mi) {
#pragma unroll
        for (int ni = 0; ni < 4; ++ni) {
            const int ncol = n0 + ni * 16 + (lane & 15);
            const float bv = BIAS ? bias[ncol] : 0.0f;
#pragma unroll
            for (int r = 0; r < 8; ++r) {
                const int mrow = m0 + mi * 16 + r + 8 * kh;
                const float v = acc[mi][ni][r] + bv;
                if (BIAS) Cbf[(size_t)mrow * N + ncol] = (__bf16)v;
                else      Cf [(size_t)mrow * N + ncol] = v;
            }
        }
    }
}

// ---------------------------------------------------------------------------
// In-place row softmax: one 256-thread block per row of 4096 f32
// wave32 shuffle reduction + cross-wave LDS reduction
// ---------------------------------------------------------------------------
__global__ __launch_bounds__(256)
void softmax_rows(float* __restrict__ data) {
    const int row = blockIdx.x;
    float* p = data + (size_t)row * SEQ_LEN;
    const int t = threadIdx.x;
    const int w = t >> 5, l = t & 31;

    float v[16];
    float mx = -3.4e38f;
#pragma unroll
    for (int i = 0; i < 16; ++i) {
        v[i] = p[t + i * 256];
        mx = fmaxf(mx, v[i]);
    }
#pragma unroll
    for (int off = 16; off > 0; off >>= 1)
        mx = fmaxf(mx, __shfl_xor(mx, off, 32));

    __shared__ float s_red[8];
    if (l == 0) s_red[w] = mx;
    __syncthreads();
    float bm = s_red[0];
#pragma unroll
    for (int i = 1; i < 8; ++i) bm = fmaxf(bm, s_red[i]);
    __syncthreads();

    float s = 0.0f;
#pragma unroll
    for (int i = 0; i < 16; ++i) {
        v[i] = __expf(v[i] - bm);
        s += v[i];
    }
#pragma unroll
    for (int off = 16; off > 0; off >>= 1)
        s += __shfl_xor(s, off, 32);
    if (l == 0) s_red[w] = s;
    __syncthreads();
    float bs = 0.0f;
#pragma unroll
    for (int i = 0; i < 8; ++i) bs += s_red[i];

    const float inv = 1.0f / bs;
#pragma unroll
    for (int i = 0; i < 16; ++i) p[t + i * 256] = v[i] * inv;
}

// ---------------------------------------------------------------------------
// Host launcher
// ---------------------------------------------------------------------------
extern "C" void kernel_launch(void* const* d_in, const int* in_sizes, int n_in,
                              void* d_out, int out_size, void* d_ws, size_t ws_size,
                              hipStream_t stream) {
    const float* out_state = (const float*)d_in[0]; // [STATE_LEN, HIDDEN]
    const float* history   = (const float*)d_in[1]; // [SEQ_LEN, HIDDEN]
    const float* W         = (const float*)d_in[2]; // [HIDDEN, HIDDEN]
    const float* b         = (const float*)d_in[3]; // [HIDDEN]
    float* out = (float*)d_out;                     // [STATE_LEN, SEQ_LEN]

    // workspace layout (bf16 copies + bf16 energies)
    char* ws = (char*)d_ws;
    size_t off = 0;
    __bf16* hist_bf  = (__bf16*)(ws + off); off += (size_t)SEQ_LEN   * HIDDEN * 2;
    __bf16* w_bf     = (__bf16*)(ws + off); off += (size_t)HIDDEN    * HIDDEN * 2;
    __bf16* state_bf = (__bf16*)(ws + off); off += (size_t)STATE_LEN * HIDDEN * 2;
    __bf16* energ_bf = (__bf16*)(ws + off); off += (size_t)SEQ_LEN   * HIDDEN * 2;
    (void)ws_size; (void)in_sizes; (void)n_in; (void)out_size;

    // 1) f32 -> bf16 conversions
    {
        int n4 = (SEQ_LEN * HIDDEN) / 4;
        cvt_f32_bf16<<<(n4 + 255) / 256, 256, 0, stream>>>(history, hist_bf, n4);
        n4 = (HIDDEN * HIDDEN) / 4;
        cvt_f32_bf16<<<(n4 + 255) / 256, 256, 0, stream>>>(W, w_bf, n4);
        n4 = (STATE_LEN * HIDDEN) / 4;
        cvt_f32_bf16<<<(n4 + 255) / 256, 256, 0, stream>>>(out_state, state_bf, n4);
    }

    // 2) energies[SEQ_LEN, HIDDEN] = history @ W^T + b   (bf16 out)
    {
        dim3 grid(HIDDEN / 256, SEQ_LEN / 128);
        gemm_nt_wmma<true><<<grid, 256, 0, stream>>>(
            hist_bf, w_bf, b, energ_bf, nullptr, SEQ_LEN, HIDDEN, HIDDEN);
    }

    // 3) scores[STATE_LEN, SEQ_LEN] = out_state @ energies^T   (f32 out -> d_out)
    {
        dim3 grid(SEQ_LEN / 256, STATE_LEN / 128);
        gemm_nt_wmma<false><<<grid, 256, 0, stream>>>(
            state_bf, energ_bf, nullptr, nullptr, out, STATE_LEN, SEQ_LEN, HIDDEN);
    }

    // 4) row softmax in place on d_out
    softmax_rows<<<STATE_LEN, 256, 0, stream>>>(out);
}
